// BioHamaModel_77421080478016
// MI455X (gfx1250) — compile-verified, matrix operand order checked by reading, not run on hardware
//
#include <hip/hip_runtime.h>
#include <cstdint>
#include <cstddef>

typedef _Float16 half_t;
typedef __attribute__((ext_vector_type(16))) _Float16 v16h;
typedef __attribute__((ext_vector_type(8)))  _Float16 v8h;
typedef __attribute__((ext_vector_type(8)))  float    v8f;
typedef __attribute__((ext_vector_type(4)))  unsigned int v4u;
typedef __attribute__((ext_vector_type(8)))  int v8i;
typedef __attribute__((ext_vector_type(4)))  int v4i;

#define DD 128
#define MM 8
#define NMEMC 16
#define BBATCH 4
#define SSEQ 512
#define VVOC 32000
#define DHH 32

// ---------------- WMMA helpers ----------------

__device__ inline v8f wmma16(v16h a, v16h b, v8f c){
  return __builtin_amdgcn_wmma_f32_16x16x32_f16(false, a, false, b, (short)0, c, false, false);
}

__device__ inline v8f zero8f(){
  v8f c;
#pragma unroll
  for(int i=0;i<8;i++) c[i]=0.f;
  return c;
}
__device__ inline v16h zero16h(){
  v16h r;
#pragma unroll
  for(int i=0;i<16;i++) r[i]=(_Float16)0.f;
  return r;
}

// Launder a uniform pointer so LLVM cannot hoist/CSE the loads fed by it
// across the 512-step recurrence loop (prevents mass VGPR spill of weight
// fragments to scratch, which round 1 asm showed in the hot loop).
__device__ inline const half_t* opaque_p(const half_t* p){
  unsigned long long u = (unsigned long long)p;
  asm volatile("" : "+s"(u));
  return (const half_t*)u;
}

// 16 contiguous f16 (one packed-fragment lane slice)
__device__ inline v16h ld16(const half_t* __restrict__ p){
  const v8h lo = *(const v8h*)p;
  const v8h hi = *(const v8h*)(p + 8);
  v16h r;
#pragma unroll
  for(int e=0;e<8;e++){ r[e]=lo[e]; r[8+e]=hi[e]; }
  return r;
}

// A-fragment (16x32) from row-major [16][ld] f16 tile.
// Lane L<16: row=L, k elems {0..7, 16..23}; L>=16: row=L-16, k {8..15, 24..31}.
__device__ inline v16h frag_rowmajor(const half_t* __restrict__ A, int ld, int kbase, int lane){
  int row = lane & 15;
  int k0  = kbase + ((lane < 16) ? 0 : 8);
  const v8h lo = *(const v8h*)(A + row*ld + k0);
  const v8h hi = *(const v8h*)(A + row*ld + k0 + 16);
  v16h r;
#pragma unroll
  for(int e=0;e<8;e++){ r[e]=lo[e]; r[8+e]=hi[e]; }
  return r;
}

// Same, but only the first 4 rows exist (B=4 batch rows); others zero.
__device__ inline v16h frag_rows4(const half_t* __restrict__ A, int ld, int kbase, int lane){
  int row = lane & 15;
  int k0  = kbase + ((lane < 16) ? 0 : 8);
  v16h r = zero16h();
  if (row < 4){
    const v8h lo = *(const v8h*)(A + row*ld + k0);
    const v8h hi = *(const v8h*)(A + row*ld + k0 + 16);
#pragma unroll
    for(int e=0;e<8;e++){ r[e]=lo[e]; r[8+e]=hi[e]; }
  }
  return r;
}

// B fragment gathered from row-major K x N source (for ctx @ v).
__device__ inline v16h frag_kn(const half_t* __restrict__ Bm, int ld, int kbase, int nbase, int lane){
  int n  = nbase + (lane & 15);
  int k0 = kbase + ((lane < 16) ? 0 : 8);
  v16h r;
#pragma unroll
  for(int e=0;e<8;e++){ r[e]   = Bm[(k0+e)*ld + n]; }
#pragma unroll
  for(int e=0;e<8;e++){ r[8+e] = Bm[(k0+16+e)*ld + n]; }
  return r;
}

// Prefetch one wave-half's 16 weight fragments (16 KB) toward the WGP.
__device__ inline void pf_weights(const half_t* __restrict__ Bp, int nhalf, int lane){
  __builtin_prefetch(Bp + (size_t)(nhalf*16)*512 + (size_t)lane*256, 0, 3);
}

// ---------------- Phase A: embed / gates / conn prefix / weight packing ----------------

__global__ void embed_kernel(const int* __restrict__ ids, const float* __restrict__ emb,
                             const float* __restrict__ pos, float* __restrict__ x){
  int tok = blockIdx.x;           // b*512 + s
  int d   = threadIdx.x;
  int s   = tok & (SSEQ-1);
  int v   = ids[tok];
  x[(size_t)tok*DD + d] = emb[(size_t)v*DD + d] + pos[(size_t)s*DD + d];
}

__global__ void gates_kernel(const float* __restrict__ x, const float* __restrict__ Wr1,
                             const float* __restrict__ br1, const float* __restrict__ Wr2,
                             const float* __restrict__ br2, float* __restrict__ gates_g){
  __shared__ float sx[DD], sh[DD], sl[MM];
  int tok = blockIdx.x, d = threadIdx.x;
  sx[d] = x[(size_t)tok*DD + d];
  __syncthreads();
  float acc = br1[d];
  for(int k=0;k<DD;k++) acc += sx[k]*Wr1[k*DD + d];
  sh[d] = fmaxf(acc, 0.f);
  __syncthreads();
  if (d < MM){
    float a = br2[d];
    for(int k=0;k<DD;k++) a += sh[k]*Wr2[k*MM + d];
    sl[d] = a;
  }
  __syncthreads();
  if (d == 0){
    float m = sl[0];
    for(int i=1;i<MM;i++) m = fmaxf(m, sl[i]);
    float s = 0.f;
    for(int i=0;i<MM;i++){ sl[i] = expf(sl[i]-m); s += sl[i]; }
    float inv = 1.f/s;
    for(int i=0;i<MM;i++) sl[i] *= inv;
  }
  __syncthreads();
  if (d < MM) gates_g[(size_t)tok*MM + d] = sl[d];
}

__global__ void conn_prefix_kernel(const float* __restrict__ conn0,
                                   const float* __restrict__ gates_g,
                                   float* __restrict__ connbuf){
  int ij = threadIdx.x;           // 0..63
  int i = ij >> 3, j = ij & 7;
  float c = conn0[ij];
  for(int t=0;t<SSEQ;t++){
    connbuf[(size_t)t*64 + ij] = c;       // conn used at step t (pre-update)
    float s = 0.f;
    for(int b=0;b<BBATCH;b++){
      size_t base = ((size_t)b*SSEQ + t)*MM;
      s += gates_g[base + i]*gates_g[base + j];
    }
    c += 0.01f * 0.25f * s;               // HEBB_LR / B
  }
}

// Pack f32 row-major [K][N] weight into WMMA B-fragment order (f16).
__global__ void pack_b_kernel(const float* __restrict__ W, half_t* __restrict__ out,
                              int K, int N){
  int idx = blockIdx.x*blockDim.x + threadIdx.x;
  int KC = K >> 5, NT = N >> 4;
  int total = NT*KC*32;
  if (idx >= total) return;
  int lane = idx & 31, f = idx >> 5;
  int kc = f % KC, nt = f / KC;
  int n  = nt*16 + (lane & 15);
  int kb = kc*32 + ((lane < 16) ? 0 : 8);
  half_t* dst = out + (size_t)idx*16;
#pragma unroll
  for(int e=0;e<8;e++) dst[e]   = (half_t)W[(size_t)(kb+e)*N + n];
#pragma unroll
  for(int e=0;e<8;e++) dst[8+e] = (half_t)W[(size_t)(kb+16+e)*N + n];
}

// ---------------- Phase B: serial recurrent scan (single persistent workgroup) ----------------

// One wave handles expert m, column half nhalf. Double-buffered B fragments,
// bias from preloaded registers, no global loads hoisted out of the t loop.
__device__ inline void expert_layer_gemm(const half_t* __restrict__ Asrc,  // [4][128] f16
                                         const half_t* __restrict__ Bp,    // opaque'd packed weights (this expert)
                                         const float* bias4,               // 4 per-lane bias values
                                         float* __restrict__ outb,         // [4][8][128]
                                         int m, int nhalf, int lane){
  v16h a[4];
#pragma unroll
  for(int kc=0;kc<4;kc++) a[kc] = frag_rows4(Asrc, DD, kc*32, lane);
  const half_t* lanebase = Bp + (size_t)(nhalf*16)*512 + (size_t)lane*16;
  v16h bf[2][4];
#pragma unroll
  for(int kc=0;kc<4;kc++) bf[0][kc] = ld16(lanebase + (size_t)kc*512);
#pragma unroll
  for(int tix=0;tix<4;tix++){
    int cur = tix & 1, nxt = cur ^ 1;
    if (tix < 3){
#pragma unroll
      for(int kc=0;kc<4;kc++) bf[nxt][kc] = ld16(lanebase + (size_t)((tix+1)*4 + kc)*512);
    }
    v8f acc = zero8f();
#pragma unroll
    for(int kc=0;kc<4;kc++) acc = wmma16(a[kc], bf[cur][kc], acc);
    int col = (nhalf*4 + tix)*16 + (lane & 15);
#pragma unroll
    for(int r=0;r<8;r++){
      int row = r + ((lane < 16) ? 0 : 8);
      if (row < 4) outb[(row*MM + m)*DD + col] = acc[r] + bias4[tix];
    }
  }
}

__device__ inline void ln_relu_to_act(const float* __restrict__ inb,   // [4][8][128]
                                      const float* g8, const float* b8, // 8 per-thread LN params (regs)
                                      half_t* __restrict__ h_act,       // [8][4][128]
                                      float* s_red, float* s_stat, int tid){
  { // 32 rows x 16 partials (512 threads exactly)
    int row = tid >> 4, p = tid & 15;
    float s0=0.f, s1=0.f;
    for(int j=0;j<8;j++){ float v = inb[row*DD + p*8 + j]; s0+=v; s1+=v*v; }
    s_red[tid] = s0; s_red[512+tid] = s1;
  }
  __syncthreads();
  if (tid < 32){
    float s0=0.f, s1=0.f;
    for(int p=0;p<16;p++){ s0 += s_red[tid*16+p]; s1 += s_red[512+tid*16+p]; }
    float m  = s0*(1.f/128.f);
    float va = s1*(1.f/128.f) - m*m;
    s_stat[tid*2] = m; s_stat[tid*2+1] = rsqrtf(va + 1e-5f);
  }
  __syncthreads();
#pragma unroll
  for(int k=0;k<8;k++){
    int i = tid + k*512;
    int b = i >> 10, m = (i >> 7) & 7, d = i & 127;
    int row = b*MM + m;
    float mn = s_stat[row*2], iv = s_stat[row*2+1];
    float v = (inb[i]-mn)*iv*g8[k] + b8[k];
    h_act[(m*4 + b)*DD + d] = (half_t)fmaxf(v, 0.f);
  }
}

__global__ __launch_bounds__(512)
void scan_kernel(const float* __restrict__ x, const float* __restrict__ gates_g,
                 const float* __restrict__ connbuf, const float* __restrict__ mem0,
                 const float* __restrict__ ln_g, const float* __restrict__ ln_b,
                 const half_t* __restrict__ eW1p, const float* __restrict__ eb1,
                 const float* __restrict__ eln1g, const float* __restrict__ eln1b,
                 const half_t* __restrict__ eW2p, const float* __restrict__ eb2,
                 const float* __restrict__ eln2g, const float* __restrict__ eln2b,
                 const half_t* __restrict__ eW3p, const float* __restrict__ eb3,
                 const half_t* __restrict__ Wmsgp,
                 float* __restrict__ out_f, half_t* __restrict__ out_h){
  extern __shared__ char smemraw[];
  float* s_mem    = (float*)smemraw;          // 4*16*128 = 8192
  float* s_cur    = s_mem    + 8192;          // 512
  float* s_comb   = s_cur    + 512;           // 512
  float* s_states = s_comb   + 512;           // 4096
  float* s_msg    = s_states + 4096;          // 4096 (also raw pre-LN expert output)
  float* s_agg    = s_msg    + 4096;          // 512
  float* s_red    = s_agg    + 512;           // 1024
  float* s_stat   = s_red    + 1024;          // 64
  float* s_sc     = s_stat   + 64;            // 64
  float* s_w      = s_sc     + 64;            // 64
  float* s_gate   = s_w      + 64;            // 32
  float* s_conn   = s_gate   + 32;            // 64
  half_t* h_comb  = (half_t*)(s_conn + 64);   // 512
  half_t* h_act   = h_comb + 512;             // 8*4*128 = 4096
  half_t* h_msg   = h_act  + 4096;            // 4*8*128 = 4096

  const int tid  = threadIdx.x;
  const int lane = tid & 31;
  const int wv   = tid >> 5;
  const int bq   = tid >> 7;                  // batch for (b,d) work split
  const int dq   = tid & 127;
  const int me    = wv & 7;                   // expert for this wave
  const int nhalf = wv >> 3;                  // column half (0..1)
  const float iscale = 0.08838834764831845f;  // 1/sqrt(128)

  // step-invariant preloads (keeps all constant params out of the hot loop)
  const float lngv = ln_g[dq], lnbv = ln_b[dq];
  float b1v[4], b2v[4], b3v[4];
#pragma unroll
  for(int tix=0;tix<4;tix++){
    int col = (nhalf*4 + tix)*16 + (lane & 15);
    b1v[tix] = eb1[me*DD + col];
    b2v[tix] = eb2[me*DD + col];
    b3v[tix] = eb3[me*DD + col];
  }
  float g1r[8], be1r[8], g2r[8], be2r[8];
#pragma unroll
  for(int k=0;k<8;k++){
    int i = tid + k*512;
    int m2 = (i >> 7) & 7, d2 = i & 127;
    g1r[k]=eln1g[m2*DD+d2]; be1r[k]=eln1b[m2*DD+d2];
    g2r[k]=eln2g[m2*DD+d2]; be2r[k]=eln2b[m2*DD+d2];
  }

  for(int i=tid;i<8192;i+=512) s_mem[i] = mem0[i & 2047];
  __syncthreads();

  for(int t=0;t<SSEQ;++t){
    // launder weight pointers each step: loads stay in-loop (L2 hits), no spills
    const half_t* W1 = opaque_p(eW1p) + (size_t)me*16384;
    const half_t* W2 = opaque_p(eW2p) + (size_t)me*16384;
    const half_t* W3 = opaque_p(eW3p) + (size_t)me*16384;
    const half_t* WM = opaque_p(Wmsgp);
    pf_weights(W1, nhalf, lane);

    s_cur[tid] = x[((size_t)bq*SSEQ + t)*DD + dq];
    if (tid < 32) s_gate[tid] = gates_g[(((size_t)(tid>>3))*SSEQ + t)*MM + (tid&7)];
    if (tid < 64) s_conn[tid] = connbuf[(size_t)t*64 + tid];
    __syncthreads();

    // memory read attention
    if (tid < 64){
      int b = tid>>4, n = tid&15;
      const float* mb = s_mem + (b*NMEMC + n)*DD;
      const float* cb = s_cur + b*DD;
      float s = 0.f;
      for(int d=0;d<DD;d++) s += cb[d]*mb[d];
      s_sc[tid] = s*iscale;
    }
    __syncthreads();
    if (tid < 4){
      float* p = s_sc + tid*16;
      float m = p[0]; for(int n=1;n<16;n++) m = fmaxf(m, p[n]);
      float s = 0.f;  for(int n=0;n<16;n++){ p[n] = expf(p[n]-m); s += p[n]; }
      float inv = 1.f/s; for(int n=0;n<16;n++) p[n] *= inv;
    }
    __syncthreads();
    { // retrieved + pre-LN combined
      float r = 0.f;
      for(int n=0;n<NMEMC;n++) r += s_sc[bq*16+n]*s_mem[(bq*NMEMC+n)*DD + dq];
      s_comb[tid] = s_cur[tid] + r;
    }
    __syncthreads();
    if (tid < 64){
      int b = tid>>4, p = tid&15;
      float s0=0.f, s1=0.f;
      for(int j=0;j<8;j++){ float v = s_comb[b*DD + p*8 + j]; s0+=v; s1+=v*v; }
      s_red[tid]=s0; s_red[512+tid]=s1;
    }
    __syncthreads();
    if (tid < 4){
      float s0=0.f, s1=0.f;
      for(int p=0;p<16;p++){ s0+=s_red[tid*16+p]; s1+=s_red[512+tid*16+p]; }
      float m = s0*(1.f/128.f), va = s1*(1.f/128.f)-m*m;
      s_stat[tid*2]=m; s_stat[tid*2+1]=rsqrtf(va+1e-5f);
    }
    __syncthreads();
    {
      float m = s_stat[bq*2], iv = s_stat[bq*2+1];
      float v = (s_comb[tid]-m)*iv*lngv + lnbv;
      s_comb[tid] = v; h_comb[tid] = (half_t)v;
    }
    __syncthreads();

    // expert MLP layer 1 (WMMA) -> raw into s_msg, then LN+relu -> h_act
    expert_layer_gemm(h_comb, W1, b1v, s_msg, me, nhalf, lane);
    pf_weights(W2, nhalf, lane);
    __syncthreads();
    ln_relu_to_act(s_msg, g1r, be1r, h_act, s_red, s_stat, tid);
    __syncthreads();
    // layer 2
    expert_layer_gemm(h_act + (size_t)me*512, W2, b2v, s_msg, me, nhalf, lane);
    pf_weights(W3, nhalf, lane);
    __syncthreads();
    ln_relu_to_act(s_msg, g2r, be2r, h_act, s_red, s_stat, tid);
    __syncthreads();
    // layer 3 -> states (f32, + bias, no LN/relu)
    expert_layer_gemm(h_act + (size_t)me*512, W3, b3v, s_states, me, nhalf, lane);
    __syncthreads();

    { // agg = gates . states (before message passing mutates states)
      float a = 0.f;
      for(int m2=0;m2<MM;m2++) a += s_gate[bq*MM+m2]*s_states[(bq*MM+m2)*DD + dq];
      s_agg[tid] = a;
    }

    // message passing: s = s + relu((conn @ (s*gates)) @ Wmsg), 2 steps
    for(int mp=0;mp<2;++mp){
      __syncthreads();
      for(int i=tid;i<4096;i+=512){
        int b = i>>10, ii = (i>>7)&7, d = i&127;
        float acc = 0.f;
        for(int j=0;j<MM;j++) acc += s_conn[ii*MM+j]*s_gate[b*MM+j]*s_states[(b*MM+j)*DD + d];
        s_msg[i] = acc; h_msg[i] = (half_t)acc;
      }
      __syncthreads();
      {
        const half_t* WMi = opaque_p(WM);
        int rt = wv & 1, ntl = wv >> 1;     // 2 row-tiles x 8 col-tiles over 16 waves
        const half_t* lanebase = WMi + (size_t)(ntl*4)*512 + (size_t)lane*16;
        v16h a[4];
#pragma unroll
        for(int kc=0;kc<4;kc++) a[kc] = frag_rowmajor(h_msg + rt*16*DD, DD, kc*32, lane);
        v8f acc = zero8f();
#pragma unroll
        for(int kc=0;kc<4;kc++) acc = wmma16(a[kc], ld16(lanebase + (size_t)kc*512), acc);
        int col = ntl*16 + (lane & 15);
#pragma unroll
        for(int r=0;r<8;r++){
          int row = rt*16 + r + ((lane<16)?0:8);   // row = b*8+m
          s_states[row*DD + col] += fmaxf(acc[r], 0.f);
        }
      }
    }
    __syncthreads();
    { // agg += mean_m s
      float s = 0.f;
      for(int m2=0;m2<MM;m2++) s += s_states[(bq*MM+m2)*DD + dq];
      s_agg[tid] += s*0.125f;
    }
    __syncthreads();
    // memory write attention
    if (tid < 64){
      int b = tid>>4, n = tid&15;
      const float* mb = s_mem + (b*NMEMC+n)*DD;
      const float* ab = s_agg + b*DD;
      float s = 0.f;
      for(int d=0;d<DD;d++) s += ab[d]*mb[d];
      s_w[tid] = s*iscale;
    }
    __syncthreads();
    if (tid < 4){
      float* p = s_w + tid*16;
      float m = p[0]; for(int n=1;n<16;n++) m = fmaxf(m, p[n]);
      float s = 0.f;  for(int n=0;n<16;n++){ p[n] = expf(p[n]-m); s += p[n]; }
      float inv = 1.f/s; for(int n=0;n<16;n++) p[n] *= inv;
    }
    __syncthreads();
    {
      float a = s_agg[tid];
      for(int n=0;n<NMEMC;n++) s_mem[(bq*NMEMC+n)*DD + dq] += s_w[bq*16+n]*a;
      size_t o = ((size_t)bq*SSEQ + t)*DD + dq;
      out_f[o] = a; out_h[o] = (half_t)a;
    }
    __syncthreads();
  }
}

// ---------------- Phase C/D: WMMA GEMMs, sparse attention, decoder ----------------

// Stage the block's shared A tile (16 rows x K f16) into LDS via the Tensor
// Data Mover (gfx1250 TDM), then all 8 waves build fragments from LDS.
__device__ inline void stage_a_tile_lds(half_t* __restrict__ sA,
                                        const half_t* __restrict__ Arow, int K,
                                        int tid, int wv){
#if defined(__has_builtin) && __has_builtin(__builtin_amdgcn_tensor_load_to_lds)
  if (wv == 0){
    unsigned long long ga = (unsigned long long)Arow;
    unsigned int w = (unsigned int)(K >> 1);     // row length/stride in 4-byte units
    v4u g0; v8i g1; v4i g2; v4i g3;
    g0[0] = 1u;                                   // count=1, user descriptor
    g0[1] = 0u;                                   // lds_addr = 0 (dynamic LDS base)
    g0[2] = (unsigned int)(ga & 0xffffffffull);
    g0[3] = (unsigned int)((ga >> 32) & 0x1ffffffull) | 0x80000000u;  // type=2 (image)
    g1[0] = (int)(2u << 16);                      // workgroup_mask=0, data_size=4B
    g1[1] = (int)(w << 16);                       // tensor_dim0[15:0]
    g1[2] = (int)(0xffffu << 16);                 // tensor_dim1[15:0] (large)
    g1[3] = (int)(w << 16);                       // tile_dim0
    g1[4] = 16;                                   // tile_dim1 = 16 rows
    g1[5] = (int)w;                               // tensor_dim0_stride (low 32)
    g1[6] = 0; g1[7] = 0;
#pragma unroll
    for(int i=0;i<4;i++){ g2[i]=0; g3[i]=0; }
#if __clang_major__ >= 23
    v8i g4;
#pragma unroll
    for(int i=0;i<8;i++) g4[i]=0;
    __builtin_amdgcn_tensor_load_to_lds(g0, g1, g2, g3, g4, 0);
#else
    __builtin_amdgcn_tensor_load_to_lds(g0, g1, g2, g3, 0);
#endif
    __builtin_amdgcn_s_wait_tensorcnt(0);
  }
  __syncthreads();
#else
  // fallback: cooperative vector copy
  int n8 = (16*K) >> 3;                           // number of v8h chunks
  for(int i=tid;i<n8;i+=blockDim.x)
    ((v8h*)sA)[i] = ((const v8h*)Arow)[i];
  __syncthreads();
#endif
}

// Generic GEMM: C[M,N] = A[M,K] (f16 row-major) * Bpacked + bias, optional relu.
__global__ __launch_bounds__(256)
void gemm_wmma_kernel(const half_t* __restrict__ A, const half_t* __restrict__ Bp,
                      const float* __restrict__ bias, float* __restrict__ Cf,
                      half_t* __restrict__ Ch, int N, int K, int tilesPerWave, int doRelu){
  extern __shared__ half_t sA[];                  // 16 x K staged A tile (<=4KB)
  int tid = threadIdx.x;
  int lane = tid & 31, wv = tid >> 5;
  int rowtile = blockIdx.y;
  int KC = K >> 5, NT = N >> 4;
  const half_t* Arow = A + (size_t)rowtile*16*K;
  stage_a_tile_lds(sA, Arow, K, tid, wv);
  v16h a[4];
#pragma unroll
  for(int kc=0;kc<4;kc++) if (kc < KC) a[kc] = frag_rowmajor(sA, K, kc*32, lane);
  int base = blockIdx.x*(8*tilesPerWave) + wv*tilesPerWave;
  for(int t=0;t<tilesPerWave;t++){
    int nt = base + t;
    if (nt >= NT) break;
    const half_t* lanebase = Bp + (size_t)(nt*KC)*512 + (size_t)lane*16;
    if (t+1 < tilesPerWave && nt+1 < NT)
      __builtin_prefetch(Bp + (size_t)((nt+1)*KC)*512 + (size_t)lane*64, 0, 3);
    v8f acc = zero8f();
    for(int kc=0;kc<KC;kc++) acc = wmma16(a[kc], ld16(lanebase + (size_t)kc*512), acc);
    int col = nt*16 + (lane & 15);
    float bv = bias ? bias[col] : 0.f;
#pragma unroll
    for(int r=0;r<8;r++){
      int row = rowtile*16 + r + ((lane<16)?0:8);
      float v = acc[r] + bv;
      if (doRelu) v = fmaxf(v, 0.f);
      size_t off = (size_t)row*N + col;
      if (Cf) Cf[off] = v;
      if (Ch) Ch[off] = (half_t)v;
    }
  }
}

// scores[bh][q][k] = (q . k) / sqrt(dh); K=32, one WMMA per tile.
__global__ __launch_bounds__(256)
void scores_kernel(const half_t* __restrict__ q_h, const half_t* __restrict__ k_h,
                   float* __restrict__ scores, float scale){
  int lane = threadIdx.x & 31, wv = threadIdx.x >> 5;
  int qtile = blockIdx.x;          // 0..31
  int bh = blockIdx.y;             // 0..15
  int b = bh >> 2, h = bh & 3;
  const half_t* qb = q_h + ((size_t)b*SSEQ)*DD + h*DHH;
  const half_t* kb = k_h + ((size_t)b*SSEQ)*DD + h*DHH;
  v16h a = frag_rowmajor(qb + (size_t)qtile*16*DD, DD, 0, lane);
  float* sc = scores + (size_t)bh*SSEQ*SSEQ;
  for(int t=0;t<4;t++){
    int nt = wv*4 + t;             // 0..31 column tiles
    v16h bf = frag_rowmajor(kb + (size_t)nt*16*DD, DD, 0, lane);  // B^T gather
    v8f acc = wmma16(a, bf, zero8f());
    int col = nt*16 + (lane & 15);
#pragma unroll
    for(int r=0;r<8;r++){
      int row = qtile*16 + r + ((lane<16)?0:8);
      sc[(size_t)row*SSEQ + col] = acc[r]*scale;
    }
  }
}

// Per-row top-k threshold (value bisection) + masked softmax -> f16 probs.
__global__ __launch_bounds__(256)
void topk_softmax_kernel(const float* __restrict__ scores, half_t* __restrict__ a_h, int kk){
  int lane = threadIdx.x & 31, wv = threadIdx.x >> 5;
  int row = blockIdx.x*8 + wv;     // 0..8191
  const float* base = scores + (size_t)row*SSEQ;
  float v[16];
#pragma unroll
  for(int i=0;i<16;i++) v[i] = base[lane + 32*i];
  float mx = v[0], mn = v[0];
#pragma unroll
  for(int i=1;i<16;i++){ mx = fmaxf(mx, v[i]); mn = fminf(mn, v[i]); }
  for(int o=16;o>0;o>>=1){ mx = fmaxf(mx, __shfl_xor(mx,o,32)); mn = fminf(mn, __shfl_xor(mn,o,32)); }
  float lo = mn, hi = mx;
  for(int it=0; it<48; ++it){
    float mid = 0.5f*(lo+hi);
    int c = 0;
#pragma unroll
    for(int i=0;i<16;i++) c += (v[i] >= mid) ? 1 : 0;
    for(int o=16;o>0;o>>=1) c += __shfl_xor(c,o,32);
    if (c >= kk) lo = mid; else hi = mid;
  }
  float thr = lo;
  float e[16]; float s = 0.f;
#pragma unroll
  for(int i=0;i<16;i++){
    float xx = (v[i] >= thr) ? expf(v[i]-mx) : 0.f;
    e[i] = xx; s += xx;
  }
  for(int o=16;o>0;o>>=1) s += __shfl_xor(s,o,32);
  float inv = 1.f/s;
  half_t* out = a_h + (size_t)row*SSEQ;
#pragma unroll
  for(int i=0;i<16;i++) out[lane + 32*i] = (half_t)(e[i]*inv);
}

// ctx[b,q,h*32+d] = sum_k a[bh,q,k] * v[b,k,h*32+d]  (K=512)
__global__ __launch_bounds__(256)
void ctx_kernel(const half_t* __restrict__ a_h, const half_t* __restrict__ v_h,
                half_t* __restrict__ ctx_h){
  int lane = threadIdx.x & 31, wv = threadIdx.x >> 5;
  int bh = blockIdx.y; int b = bh >> 2, h = bh & 3;
  int qtile = blockIdx.x*8 + wv;   // 0..31
  const half_t* ab = a_h + (size_t)bh*SSEQ*SSEQ + (size_t)qtile*16*SSEQ;
  const half_t* vb = v_h + ((size_t)b*SSEQ)*DD + h*DHH;
  v8f acc0 = zero8f(), acc1 = zero8f();
  for(int kc=0;kc<16;kc++){
    v16h a  = frag_rowmajor(ab, SSEQ, kc*32, lane);
    v16h b0 = frag_kn(vb, DD, kc*32, 0,  lane);
    v16h b1 = frag_kn(vb, DD, kc*32, 16, lane);
    acc0 = wmma16(a, b0, acc0);
    acc1 = wmma16(a, b1, acc1);
  }
  int col = lane & 15;
#pragma unroll
  for(int r=0;r<8;r++){
    int row = qtile*16 + r + ((lane<16)?0:8);
    size_t o = ((size_t)b*SSEQ + row)*DD + h*DHH;
    ctx_h[o + col]      = (half_t)acc0[r];
    ctx_h[o + 16 + col] = (half_t)acc1[r];
  }
}

// ---------------- host-side orchestration ----------------

extern "C" void kernel_launch(void* const* d_in, const int* in_sizes, int n_in,
                              void* d_out, int out_size, void* d_ws, size_t ws_size,
                              hipStream_t stream){
  (void)in_sizes; (void)n_in; (void)out_size; (void)ws_size;
  const int*   ids   = (const int*)  d_in[0];
  const float* emb   = (const float*)d_in[2];
  const float* pos   = (const float*)d_in[3];
  const float* Wr1   = (const float*)d_in[4];
  const float* br1   = (const float*)d_in[5];
  const float* Wr2   = (const float*)d_in[6];
  const float* br2   = (const float*)d_in[7];
  const float* ln_g  = (const float*)d_in[8];
  const float* ln_b  = (const float*)d_in[9];
  const float* eW1   = (const float*)d_in[10];
  const float* eb1   = (const float*)d_in[11];
  const float* eln1g = (const float*)d_in[12];
  const float* eln1b = (const float*)d_in[13];
  const float* eW2   = (const float*)d_in[14];
  const float* eb2   = (const float*)d_in[15];
  const float* eln2g = (const float*)d_in[16];
  const float* eln2b = (const float*)d_in[17];
  const float* eW3   = (const float*)d_in[18];
  const float* eb3   = (const float*)d_in[19];
  const float* mem0  = (const float*)d_in[20];
  const float* Wmsg  = (const float*)d_in[21];
  const float* conn0 = (const float*)d_in[22];
  const float* Wq    = (const float*)d_in[23];
  const float* Wk    = (const float*)d_in[24];
  const float* Wv    = (const float*)d_in[25];
  const float* Wo    = (const float*)d_in[26];
  const float* Wd1   = (const float*)d_in[27];
  const float* bd1   = (const float*)d_in[28];
  const float* Wd2   = (const float*)d_in[29];
  const float* bd2   = (const float*)d_in[30];

  char* p = (char*)d_ws;
  auto carve = [&](size_t bytes)->char*{
    char* r = p; p += (bytes + 255) & ~(size_t)255; return r;
  };
  float*  x_f    = (float*) carve(2048ull*128*4);
  float*  gatesb = (float*) carve(2048ull*8*4);
  float*  connb  = (float*) carve(512ull*64*4);
  half_t* eW1p   = (half_t*)carve(8ull*16384*2);
  half_t* eW2p   = (half_t*)carve(8ull*16384*2);
  half_t* eW3p   = (half_t*)carve(8ull*16384*2);
  half_t* Wmsgp  = (half_t*)carve(16384ull*2);
  half_t* Wqp    = (half_t*)carve(16384ull*2);
  half_t* Wkp    = (half_t*)carve(16384ull*2);
  half_t* Wvp    = (half_t*)carve(16384ull*2);
  half_t* Wop    = (half_t*)carve(16384ull*2);
  half_t* Wd1p   = (half_t*)carve(16384ull*2);
  half_t* Wd2p   = (half_t*)carve(2000ull*4*32*16*2);
  float*  out_f  = (float*) carve(2048ull*128*4);
  half_t* out_h  = (half_t*)carve(2048ull*128*2);
  half_t* q_h    = (half_t*)carve(2048ull*128*2);
  half_t* k_h    = (half_t*)carve(2048ull*128*2);
  half_t* v_h    = (half_t*)carve(2048ull*128*2);
  float*  sc_f   = (float*) carve(16ull*512*512*4);
  half_t* a_h    = (half_t*)carve(16ull*512*512*2);
  half_t* ctx_h  = (half_t*)carve(2048ull*128*2);
  half_t* out2_h = (half_t*)carve(2048ull*128*2);
  half_t* h1_h   = (half_t*)carve(2048ull*128*2);

  // Phase A: parallel precompute
  embed_kernel<<<2048, 128, 0, stream>>>(ids, emb, pos, x_f);
  gates_kernel<<<2048, 128, 0, stream>>>(x_f, Wr1, br1, Wr2, br2, gatesb);
  conn_prefix_kernel<<<1, 64, 0, stream>>>(conn0, gatesb, connb);
  for(int m=0;m<8;m++){
    pack_b_kernel<<<4, 256, 0, stream>>>(eW1 + (size_t)m*16384, eW1p + (size_t)m*16384, 128, 128);
    pack_b_kernel<<<4, 256, 0, stream>>>(eW2 + (size_t)m*16384, eW2p + (size_t)m*16384, 128, 128);
    pack_b_kernel<<<4, 256, 0, stream>>>(eW3 + (size_t)m*16384, eW3p + (size_t)m*16384, 128, 128);
  }
  pack_b_kernel<<<4, 256, 0, stream>>>(Wmsg, Wmsgp, 128, 128);
  pack_b_kernel<<<4, 256, 0, stream>>>(Wq,   Wqp,   128, 128);
  pack_b_kernel<<<4, 256, 0, stream>>>(Wk,   Wkp,   128, 128);
  pack_b_kernel<<<4, 256, 0, stream>>>(Wv,   Wvp,   128, 128);
  pack_b_kernel<<<4, 256, 0, stream>>>(Wo,   Wop,   128, 128);
  pack_b_kernel<<<4, 256, 0, stream>>>(Wd1,  Wd1p,  128, 128);
  pack_b_kernel<<<1000, 256, 0, stream>>>(Wd2, Wd2p, 128, 32000);

  // Phase B: serial scan (single persistent workgroup, WMMA inner loops)
  size_t smem = (size_t)19232*4 + (size_t)8704*2;   // 94,336 bytes (< 320KB/WGP)
  scan_kernel<<<1, 512, smem, stream>>>(x_f, gatesb, connb, mem0, ln_g, ln_b,
      eW1p, eb1, eln1g, eln1b, eW2p, eb2, eln2g, eln2b, eW3p, eb3, Wmsgp,
      out_f, out_h);

  // Phase C: sparse multi-head attention (gemm kernels use 4KB dynamic LDS for TDM A staging)
  size_t asmem = 16*128*2;
  gemm_wmma_kernel<<<dim3(1,128), 256, asmem, stream>>>(out_h, Wqp, nullptr, nullptr, q_h, 128, 128, 1, 0);
  gemm_wmma_kernel<<<dim3(1,128), 256, asmem, stream>>>(out_h, Wkp, nullptr, nullptr, k_h, 128, 128, 1, 0);
  gemm_wmma_kernel<<<dim3(1,128), 256, asmem, stream>>>(out_h, Wvp, nullptr, nullptr, v_h, 128, 128, 1, 0);
  scores_kernel<<<dim3(32,16), 256, 0, stream>>>(q_h, k_h, sc_f, 0.17677669529663687f); // 1/sqrt(32)
  topk_softmax_kernel<<<1024, 256, 0, stream>>>(sc_f, a_h, 153);                        // int(0.3*512)
  ctx_kernel<<<dim3(4,16), 256, 0, stream>>>(a_h, v_h, ctx_h);
  gemm_wmma_kernel<<<dim3(1,128), 256, asmem, stream>>>(ctx_h, Wop, nullptr, nullptr, out2_h, 128, 128, 1, 0);

  // Phase D: decoder
  gemm_wmma_kernel<<<dim3(1,128), 256, asmem, stream>>>(out2_h, Wd1p, bd1, nullptr, h1_h, 128, 128, 1, 1);
  gemm_wmma_kernel<<<dim3(63,128), 256, asmem, stream>>>(h1_h, Wd2p, bd2, (float*)d_out, nullptr, 32000, 128, 4, 0);
}
